// Net_14963666059952
// MI455X (gfx1250) — matrix operand order repaired and checked by reading, not compiled, not run on hardware
//
#include <hip/hip_runtime.h>
#include <stdint.h>
#include <stddef.h>

#define N_NODES 8192
#define N_EDGES 49152
#define HIDC    64
#define OUTC    5

typedef __attribute__((ext_vector_type(16))) __bf16 v16bf;
typedef __attribute__((ext_vector_type(8)))  float  v8f;
typedef __attribute__((ext_vector_type(4)))  unsigned int v4u;
typedef __attribute__((ext_vector_type(8)))  int   v8i;
typedef __attribute__((ext_vector_type(4)))  int   v4i;

#if defined(__has_builtin)
#if __has_builtin(__builtin_amdgcn_tensor_load_to_lds) && __has_builtin(__builtin_amdgcn_s_wait_tensorcnt)
#define HAVE_TDM 1
#endif
#endif
#ifndef HAVE_TDM
#define HAVE_TDM 0
#endif

struct alignas(16) U4 { unsigned int x, y, z, w; };

union Frag {
  v16bf v;
  unsigned short u[16];
  U4 q[2];
};

__device__ __forceinline__ unsigned short f2bf(float f) {
  union { float f; unsigned int u; } c; c.f = f;
  return (unsigned short)(c.u >> 16);
}

__device__ __forceinline__ v8f zero8() {
  v8f z;
  #pragma unroll
  for (int i = 0; i < 8; ++i) z[i] = 0.0f;
  return z;
}

// --- CDNA5 async global->LDS copy (ASYNCcnt-tracked), 16B per lane (fallback)
__device__ __forceinline__ void async_ldst_b128(unsigned lds_off, const void* gptr) {
  asm volatile("global_load_async_to_lds_b128 %0, %1, off"
               :: "v"(lds_off), "v"(gptr) : "memory");
}
__device__ __forceinline__ void wait_async_le4() {
  asm volatile("s_wait_asynccnt 0x4" ::: "memory");
}
__device__ __forceinline__ void wait_async_0() {
  asm volatile("s_wait_asynccnt 0x0" ::: "memory");
}

#if HAVE_TDM
// --- Tensor Data Mover: one-descriptor DMA of a 2D bf16 tile into LDS.
//   tile: 128 elements (256B) x 64 rows, global row stride = Delems halves.
//   pad_enable inserts 4 DWORDs (16B) after every 64 DWORDs (256B) stored,
//   reproducing the LDP = 128+8 halves padded LDS row layout.
__device__ __forceinline__ void tdm_stage(unsigned lds_addr,
                                          const unsigned short* g_src,
                                          long long Delems) {
  const unsigned long long ga = (unsigned long long)(uintptr_t)g_src;
  v4u g0;
  g0[0] = 1u;                                        // count=1, no gather
  g0[1] = lds_addr;                                  // LDS byte address
  g0[2] = (unsigned)(ga & 0xFFFFFFFFu);              // global addr [31:0]
  g0[3] = (unsigned)((ga >> 32) & 0x01FFFFFFu)       // global addr [56:32]
        | (2u << 30);                                // type = 2 ("image")
  const unsigned long long stride = (unsigned long long)Delems;  // dim0 stride
  v8i g1;
  g1[0] = (int)((1u << 16)                           // data_size = 2 bytes
        | (1u << 20)                                 // pad_enable
        | (5u << 22)                                 // pad_interval: 64 DWORDs
        | (3u << 25));                               // pad_amount: 4 DWORDs
  g1[1] = (int)(0xFFFFu << 16);                      // tensor_dim0 lo16 (huge)
  g1[2] = (int)(0x7FFFu | (0xFFFFu << 16));          // td0 hi16 | td1 lo16
  g1[3] = (int)(0x7FFFu | (128u << 16));             // td1 hi16 | tile_dim0=128
  g1[4] = 64;                                        // tile_dim1=64, tile_dim2=0
  g1[5] = (int)(unsigned)(stride & 0xFFFFFFFFu);     // dim0 stride lo32
  g1[6] = (int)((stride >> 32) & 0xFFFFu);           // dim0 stride hi16
  g1[7] = 0;
  v4i g2 = {0, 0, 0, 0};
  v4i g3 = {0, 0, 0, 0};
#if __clang_major__ >= 23
  v8i g4 = {0, 0, 0, 0, 0, 0, 0, 0};
  __builtin_amdgcn_tensor_load_to_lds(g0, g1, g2, g3, g4, 0);
#else
  __builtin_amdgcn_tensor_load_to_lds(g0, g1, g2, g3, 0);
#endif
}
#endif  // HAVE_TDM

// ---------------------------------------------------------------------------
// One-time weight prep: W2 (D x D f32, row-major [k][c]) -> w2t (bf16,
// transposed: w2t[c*D + k]).  Makes each (kc, i) staging slice 64 rows of
// 256 contiguous bytes => TDM/async-copyable, halves L2-resident footprint.
// ---------------------------------------------------------------------------
__global__ __launch_bounds__(256)
void transpose_bf16_kernel(const float* __restrict__ w,
                           unsigned short* __restrict__ wt, int D) {
  __shared__ float tile[32][33];
  const int bx = blockIdx.x * 32;
  const int by = blockIdx.y * 32;
  const int tx = threadIdx.x & 31;
  const int ty = threadIdx.x >> 5;
  #pragma unroll
  for (int r = ty; r < 32; r += 8)
    tile[r][tx] = w[(size_t)(by + r) * D + (bx + tx)];
  __syncthreads();
  #pragma unroll
  for (int r = ty; r < 32; r += 8)
    wt[(size_t)(bx + r) * D + (by + tx)] = f2bf(tile[tx][r]);
}

// ---------------------------------------------------------------------------
// Fused NNConv message kernel (128 edges / workgroup, 8 waves x 16 edges).
//   H1 = relu(ea @ W1 + b1)  via WMMA (K padded 4->32), kept in LDS as bf16.
//   for i in [0, IN_C):  P = H1 @ W2[:, i*64 : +64] (WMMA, K chunked by 128)
//                        msg[e,:] += xs[e,i] * P[e,:]
// ASYNCB: stage W2 slices from pre-transposed bf16 via TDM (or per-lane async
//         global->LDS), double-buffered.  Else: VALU convert from f32.
// ---------------------------------------------------------------------------
template<int IN_C, bool ASYNCB>
__global__ __launch_bounds__(256)
void fused_msg_kernel(const float* __restrict__ ea,            // E x 4
                      const float* __restrict__ w1,            // 4 x D
                      const float* __restrict__ b1,            // D
                      const float* __restrict__ w2,            // D x D f32
                      const unsigned short* __restrict__ w2t,  // D x D bf16 (T)
                      const float* __restrict__ xs,            // E x IN_C
                      float* __restrict__ msg)                 // E x 64
{
  constexpr int D   = IN_C * 64;
  constexpr int KC  = 128;
  constexpr int LDP = KC + 8;
  constexpr int NCHUNK = D / KC;
  constexpr int BSZ = 64 * LDP;

  extern __shared__ __align__(16) unsigned char smem[];
  unsigned short* sH1 = (unsigned short*)smem;            // 128 * LDP
  unsigned short* sB  = sH1 + 128 * LDP;                  // 2 * BSZ
  float*          sEA = (float*)(sB + 2 * BSZ);           // 128 * 4

  const int tid  = threadIdx.x;
  const int lane = tid & 31;
  const int wv   = tid >> 5;
  const int eb   = blockIdx.x * 128;
  const int erow = lane & 15;
  const int hi   = (lane >> 4) & 1;
  const int ew   = eb + wv * 16;

  if (tid < 128) {
    #pragma unroll
    for (int j = 0; j < 4; ++j) sEA[tid * 4 + j] = ea[(size_t)(eb + tid) * 4 + j];
  }
  __syncthreads();

  Frag Aea;
  #pragma unroll
  for (int j = 0; j < 16; ++j) Aea.u[j] = 0;
  if (hi == 0) {
    #pragma unroll
    for (int j = 0; j < 4; ++j)
      Aea.u[j] = f2bf(sEA[(wv * 16 + erow) * 4 + j]);
  }

  v8f acc[4];
  #pragma unroll
  for (int nt = 0; nt < 4; ++nt) acc[nt] = zero8();

  #pragma unroll 1
  for (int kc = 0; kc < NCHUNK; ++kc) {
    // ---- H1 chunk (this wave's 16 edges x KC cols) via WMMA ----
    #pragma unroll 1
    for (int nt = 0; nt < KC / 16; ++nt) {
      const int col = kc * KC + nt * 16 + erow;
      Frag Bw1;
      #pragma unroll
      for (int j = 0; j < 16; ++j) Bw1.u[j] = 0;
      if (hi == 0) {
        #pragma unroll
        for (int j = 0; j < 4; ++j)
          Bw1.u[j] = f2bf(w1[(size_t)j * D + col]);
      }
      v8f c = zero8();
      c = __builtin_amdgcn_wmma_f32_16x16x32_bf16(false, Aea.v, false, Bw1.v,
                                                  (short)0, c, false, false);
      const float bias = b1[col];
      #pragma unroll
      for (int r = 0; r < 8; ++r) {
        const int row = r + hi * 8;
        float v = c[r] + bias;
        v = v > 0.0f ? v : 0.0f;
        sH1[(wv * 16 + row) * LDP + (nt * 16 + erow)] = f2bf(v);
      }
    }
    __syncthreads();

    // Prologue: stage slice i=0 into buffer 0.
    if constexpr (ASYNCB) {
#if HAVE_TDM
      if (wv == 0)
        tdm_stage((unsigned)(uintptr_t)sB, w2t + (size_t)kc * KC, D);
#else
      const unsigned short* sb0 = w2t + (size_t)kc * KC;
      #pragma unroll
      for (int it = 0; it < 4; ++it) {
        const int t = it * 256 + tid;
        const int o = t >> 4;
        const int q = t & 15;
        async_ldst_b128((unsigned)(uintptr_t)(sB + o * LDP + q * 8),
                        (const void*)(sb0 + (size_t)o * D + q * 8));
      }
#endif
    }

    #pragma unroll 1
    for (int i = 0; i < IN_C; ++i) {
      unsigned short* bcur;
      if constexpr (ASYNCB) {
        bcur = sB + (i & 1) * BSZ;
#if HAVE_TDM
        if (wv == 0) {
          if (i + 1 < IN_C) {
            tdm_stage((unsigned)(uintptr_t)(sB + ((i + 1) & 1) * BSZ),
                      w2t + ((size_t)(i + 1) * 64) * D + (size_t)kc * KC, D);
            __builtin_amdgcn_s_wait_tensorcnt((short)1);  // per-wave, in-order
          } else {
            __builtin_amdgcn_s_wait_tensorcnt((short)0);
          }
        }
        __syncthreads();
#else
        if (i + 1 < IN_C) {
          unsigned short* bnxt = sB + ((i + 1) & 1) * BSZ;
          const unsigned short* g = w2t + ((size_t)(i + 1) * 64) * D + (size_t)kc * KC;
          #pragma unroll
          for (int it = 0; it < 4; ++it) {
            const int t = it * 256 + tid;
            const int o = t >> 4;
            const int q = t & 15;
            async_ldst_b128((unsigned)(uintptr_t)(bnxt + o * LDP + q * 8),
                            (const void*)(g + (size_t)o * D + q * 8));
          }
          wait_async_le4();
        } else {
          wait_async_0();
        }
        __syncthreads();
#endif
      } else {
        bcur = sB;
        __syncthreads();
        #pragma unroll 1
        for (int t = tid; t < KC * 64; t += 256) {
          const int k = t >> 6;
          const int o = t & 63;
          bcur[o * LDP + k] = f2bf(w2[(size_t)(kc * KC + k) * D + (size_t)i * 64 + o]);
        }
        __syncthreads();
      }

      v8f P[4];
      #pragma unroll
      for (int nt = 0; nt < 4; ++nt) P[nt] = zero8();

      #pragma unroll 1
      for (int ks = 0; ks < KC / 32; ++ks) {
        Frag A;
        {
          const unsigned short* base =
              &sH1[(wv * 16 + erow) * LDP + ks * 32 + hi * 8];
          A.q[0] = *reinterpret_cast<const U4*>(base);
          A.q[1] = *reinterpret_cast<const U4*>(base + 16);
        }
        #pragma unroll
        for (int nt = 0; nt < 4; ++nt) {
          Frag B;
          const unsigned short* bb =
              &bcur[(nt * 16 + erow) * LDP + ks * 32 + hi * 8];
          B.q[0] = *reinterpret_cast<const U4*>(bb);
          B.q[1] = *reinterpret_cast<const U4*>(bb + 16);
          P[nt] = __builtin_amdgcn_wmma_f32_16x16x32_bf16(false, A.v, false, B.v,
                                                          (short)0, P[nt], false, false);
        }
      }

      #pragma unroll
      for (int r = 0; r < 8; ++r) {
        const int row = r + hi * 8;
        const float s = xs[(size_t)(ew + row) * IN_C + i];
        #pragma unroll
        for (int nt = 0; nt < 4; ++nt) acc[nt][r] += s * P[nt][r];
      }
      __syncthreads();  // all readers done before buffer is overwritten
    }
    __syncthreads();    // sH1 reused next kc
  }

  #pragma unroll
  for (int nt = 0; nt < 4; ++nt) {
    #pragma unroll
    for (int r = 0; r < 8; ++r) {
      const int row = r + hi * 8;
      msg[(size_t)(ew + row) * 64 + nt * 16 + erow] = acc[nt][r];
    }
  }
}

// ---------------------------------------------------------------------------
// Node update: hn = act(agg/max(cnt,1) + h @ root + bias), WMMA 16x64xIN_C.
// ---------------------------------------------------------------------------
template<int IN_C, bool RELU>
__global__ __launch_bounds__(256)
void node_update_kernel(const float* __restrict__ h,
                        const float* __restrict__ root,
                        const float* __restrict__ bias,
                        const float* __restrict__ agg,
                        const float* __restrict__ cnt,
                        float* __restrict__ hn) {
  const int lane = threadIdx.x & 31;
  const int wv   = threadIdx.x >> 5;
  const int nb   = (blockIdx.x * 8 + wv) * 16;
  const int r16  = lane & 15;
  const int hi   = (lane >> 4) & 1;
  constexpr int KS = (IN_C + 31) / 32;

  v8f C[4];
  #pragma unroll
  for (int nt = 0; nt < 4; ++nt) C[nt] = zero8();

  #pragma unroll
  for (int ks = 0; ks < KS; ++ks) {
    Frag A;
    #pragma unroll
    for (int j = 0; j < 8; ++j) {
      const int k0 = ks * 32 + hi * 8 + j;
      const int k1 = k0 + 16;
      A.u[j]     = (k0 < IN_C) ? f2bf(h[(size_t)(nb + r16) * IN_C + k0]) : (unsigned short)0;
      A.u[8 + j] = (k1 < IN_C) ? f2bf(h[(size_t)(nb + r16) * IN_C + k1]) : (unsigned short)0;
    }
    #pragma unroll
    for (int nt = 0; nt < 4; ++nt) {
      const int o = nt * 16 + r16;
      Frag B;
      #pragma unroll
      for (int j = 0; j < 16; ++j) {
        const int k = ks * 32 + hi * 8 + (j < 8 ? j : j + 8);
        B.u[j] = (k < IN_C) ? f2bf(root[(size_t)k * 64 + o]) : (unsigned short)0;
      }
      C[nt] = __builtin_amdgcn_wmma_f32_16x16x32_bf16(false, A.v, false, B.v,
                                                      (short)0, C[nt], false, false);
    }
  }

  #pragma unroll
  for (int nt = 0; nt < 4; ++nt) {
    const int o = nt * 16 + r16;
    #pragma unroll
    for (int r = 0; r < 8; ++r) {
      const int node = nb + r + hi * 8;
      float cn = cnt[node];
      cn = cn > 1.0f ? cn : 1.0f;
      float v = C[nt][r] + agg[(size_t)node * 64 + o] / cn + bias[o];
      if (RELU) v = v > 0.0f ? v : 0.0f;
      hn[(size_t)node * 64 + o] = v;
    }
  }
}

// ---------------------------------------------------------------------------
template<int C>
__global__ void gather_kernel(const float* __restrict__ h,
                              const int* __restrict__ src,
                              float* __restrict__ xs) {
  const int idx = blockIdx.x * blockDim.x + threadIdx.x;
  if (idx >= N_EDGES * C) return;
  const int e = idx / C, c = idx % C;
  xs[idx] = h[(size_t)src[e] * C + c];
}

__global__ void zero_kernel(float* __restrict__ p, int n) {
  const int i = blockIdx.x * blockDim.x + threadIdx.x;
  if (i < n) p[i] = 0.0f;
}

__global__ void count_kernel(const int* __restrict__ tgt, float* __restrict__ cnt) {
  const int e = blockIdx.x * blockDim.x + threadIdx.x;
  if (e < N_EDGES) atomicAdd(&cnt[tgt[e]], 1.0f);
}

__global__ void scatter_kernel(const float* __restrict__ msg,
                               const int* __restrict__ tgt,
                               float* __restrict__ agg) {
  const int idx = blockIdx.x * blockDim.x + threadIdx.x;
  if (idx >= N_EDGES * 64) return;
  const int e = idx >> 6, c = idx & 63;
  atomicAdd(&agg[(size_t)tgt[e] * 64 + c], msg[idx]);
}

__global__ void out_kernel(const float* __restrict__ h,
                           const float* __restrict__ w,
                           const float* __restrict__ b,
                           float* __restrict__ out) {
  const int n = blockIdx.x * blockDim.x + threadIdx.x;
  if (n >= N_NODES) return;
  float a[OUTC];
  #pragma unroll
  for (int o = 0; o < OUTC; ++o) a[o] = b[o];
  #pragma unroll 4
  for (int k = 0; k < HIDC; ++k) {
    const float hv = h[(size_t)n * HIDC + k];
    #pragma unroll
    for (int o = 0; o < OUTC; ++o) a[o] += hv * w[k * OUTC + o];
  }
  #pragma unroll
  for (int o = 0; o < OUTC; ++o) out[(size_t)n * OUTC + o] = a[o];
}

// ---------------------------------------------------------------------------
extern "C" void kernel_launch(void* const* d_in, const int* in_sizes, int n_in,
                              void* d_out, int out_size, void* d_ws, size_t ws_size,
                              hipStream_t stream) {
  const float* x    = (const float*)d_in[0];
  const int*   eidx = (const int*)d_in[1];
  const float* ea   = (const float*)d_in[2];
  const float* e1w1 = (const float*)d_in[3];
  const float* e1b1 = (const float*)d_in[4];
  const float* e1w2 = (const float*)d_in[5];
  const float* root1= (const float*)d_in[7];
  const float* bias1= (const float*)d_in[8];
  const float* e2w1 = (const float*)d_in[9];
  const float* e2b1 = (const float*)d_in[10];
  const float* e2w2 = (const float*)d_in[11];
  const float* root2= (const float*)d_in[13];
  const float* bias2= (const float*)d_in[14];
  const float* e3w1 = (const float*)d_in[15];
  const float* e3b1 = (const float*)d_in[16];
  const float* e3w2 = (const float*)d_in[17];
  const float* root3= (const float*)d_in[19];
  const float* bias3= (const float*)d_in[20];
  const float* loutw= (const float*)d_in[21];
  const float* loutb= (const float*)d_in[22];

  const int* src = eidx;
  const int* tgt = eidx + N_EDGES;

  uintptr_t p = (uintptr_t)d_ws;
  auto take = [&](size_t bytes) -> void* {
    void* r = (void*)p;
    p += (bytes + 255) & ~(size_t)255;
    return r;
  };
  float* xs  = (float*)take((size_t)N_EDGES * 64 * 4);
  float* msg = (float*)take((size_t)N_EDGES * 64 * 4);
  float* agg = (float*)take((size_t)N_NODES * 64 * 4);
  float* cnt = (float*)take((size_t)N_NODES * 4);
  float* hA  = (float*)take((size_t)N_NODES * 64 * 4);
  float* hB  = (float*)take((size_t)N_NODES * 64 * 4);
  unsigned short* w2t1 = (unsigned short*)take((size_t)512 * 512 * 2);
  unsigned short* w2t2 = (unsigned short*)take((size_t)4096 * 4096 * 2);
  unsigned short* w2t3 = (unsigned short*)take((size_t)4096 * 4096 * 2);
  const bool use_async = (p - (uintptr_t)d_ws) <= ws_size;

  const int T = 256;
  const size_t SMEM = (size_t)(128 * 136 + 2 * 64 * 136) * 2 + 128 * 4 * 4;  // 71680B

  zero_kernel<<<(N_NODES + T - 1) / T, T, 0, stream>>>(cnt, N_NODES);
  count_kernel<<<(N_EDGES + T - 1) / T, T, 0, stream>>>(tgt, cnt);

  if (use_async) {
    dim3 blk(256);
    transpose_bf16_kernel<<<dim3(512 / 32, 512 / 32), blk, 0, stream>>>(e1w2, w2t1, 512);
    transpose_bf16_kernel<<<dim3(4096 / 32, 4096 / 32), blk, 0, stream>>>(e2w2, w2t2, 4096);
    transpose_bf16_kernel<<<dim3(4096 / 32, 4096 / 32), blk, 0, stream>>>(e3w2, w2t3, 4096);
  }

  // ---- layer 1 (IN_C = 8, D = 512) ----
  gather_kernel<8><<<(N_EDGES * 8 + T - 1) / T, T, 0, stream>>>(x, src, xs);
  zero_kernel<<<(N_NODES * 64 + T - 1) / T, T, 0, stream>>>(agg, N_NODES * 64);
  if (use_async)
    fused_msg_kernel<8, true><<<N_EDGES / 128, T, SMEM, stream>>>(ea, e1w1, e1b1, e1w2, w2t1, xs, msg);
  else
    fused_msg_kernel<8, false><<<N_EDGES / 128, T, SMEM, stream>>>(ea, e1w1, e1b1, e1w2, nullptr, xs, msg);
  scatter_kernel<<<(N_EDGES * 64 + T - 1) / T, T, 0, stream>>>(msg, tgt, agg);
  node_update_kernel<8, true><<<N_NODES / 128, T, 0, stream>>>(x, root1, bias1, agg, cnt, hA);

  // ---- layer 2 (IN_C = 64, D = 4096) ----
  gather_kernel<64><<<(N_EDGES * 64 + T - 1) / T, T, 0, stream>>>(hA, src, xs);
  zero_kernel<<<(N_NODES * 64 + T - 1) / T, T, 0, stream>>>(agg, N_NODES * 64);
  if (use_async)
    fused_msg_kernel<64, true><<<N_EDGES / 128, T, SMEM, stream>>>(ea, e2w1, e2b1, e2w2, w2t2, xs, msg);
  else
    fused_msg_kernel<64, false><<<N_EDGES / 128, T, SMEM, stream>>>(ea, e2w1, e2b1, e2w2, nullptr, xs, msg);
  scatter_kernel<<<(N_EDGES * 64 + T - 1) / T, T, 0, stream>>>(msg, tgt, agg);
  node_update_kernel<64, true><<<N_NODES / 128, T, 0, stream>>>(hA, root2, bias2, agg, cnt, hB);

  // ---- layer 3 (IN_C = 64, D = 4096) ----
  gather_kernel<64><<<(N_EDGES * 64 + T - 1) / T, T, 0, stream>>>(hB, src, xs);
  zero_kernel<<<(N_NODES * 64 + T - 1) / T, T, 0, stream>>>(agg, N_NODES * 64);
  if (use_async)
    fused_msg_kernel<64, true><<<N_EDGES / 128, T, SMEM, stream>>>(ea, e3w1, e3b1, e3w2, w2t3, xs, msg);
  else
    fused_msg_kernel<64, false><<<N_EDGES / 128, T, SMEM, stream>>>(ea, e3w1, e3b1, e3w2, nullptr, xs, msg);
  scatter_kernel<<<(N_EDGES * 64 + T - 1) / T, T, 0, stream>>>(msg, tgt, agg);
  node_update_kernel<64, true><<<N_NODES / 128, T, 0, stream>>>(hB, root3, bias3, agg, cnt, hA);

  // ---- output projection ----
  out_kernel<<<(N_NODES + T - 1) / T, T, 0, stream>>>(hA, loutw, loutb, (float*)d_out);
}
